// PointerNetwork_88991722373670
// MI455X (gfx1250) — compile-verified, matrix operand order checked by reading, not run on hardware
//
#include <hip/hip_runtime.h>
#include <hip/hip_bf16.h>

// Pointer network (encoder LSTM + Bahdanau attention decoder) for gfx1250.
// Recurrent GEMMs run through v_wmma_f32_16x16x32_bf16 (bf16 operands, fp32
// accumulate); LSTM state h/c stays fp32. enc_out kept bf16 to halve L2
// traffic in the attention context phase.

#define Bsz 512
#define Ssz 128
#define Hsz 256
#define Usz 10
#define NG  1024   // 4*H

typedef __attribute__((ext_vector_type(16))) __bf16 v16bf;
typedef __attribute__((ext_vector_type(8)))  float  v8f;

union FragBF { unsigned int u[8]; v16bf v; };

__device__ __forceinline__ unsigned short f2bf(float f) {
  unsigned int u = __float_as_uint(f);
  unsigned int r = u + 0x7FFFu + ((u >> 16) & 1u);   // round-to-nearest-even
  return (unsigned short)(r >> 16);
}
__device__ __forceinline__ float bf2f(unsigned short s) {
  return __uint_as_float(((unsigned int)s) << 16);
}
__device__ __forceinline__ float sigmoidf(float x) {
  return 1.0f / (1.0f + __expf(-x));
}

// ---------------------------------------------------------------------------
// Weight conversion: Whh_e -> bf16 [1024 x 256]; decoder combined weight
// Wdec[n, k] = (k < 256) ? W_ih_d[n, k] : W_hh_d[n, k-256]   -> bf16 [1024 x 512]
// ---------------------------------------------------------------------------
__global__ __launch_bounds__(256)
void ptr_convert_weights(const float* __restrict__ Whh_e,
                         const float* __restrict__ W_ih_d,   // [4H, H+1]
                         const float* __restrict__ W_hh_d,   // [4H, H]
                         unsigned short* __restrict__ w_enc, // [4H, 256] bf16
                         unsigned short* __restrict__ w_dec) // [4H, 512] bf16
{
  int i = blockIdx.x * 256 + threadIdx.x;
  if (i < NG * Hsz) w_enc[i] = f2bf(Whh_e[i]);
  if (i < NG * 512) {
    int n = i >> 9, k = i & 511;
    float v = (k < Hsz) ? W_ih_d[n * (Hsz + 1) + k]
                        : W_hh_d[n * Hsz + (k - Hsz)];
    w_dec[i] = f2bf(v);
  }
}

// ---------------------------------------------------------------------------
// WMMA GEMM: out[M x N] = A[M x K](bf16, row-major) * W[N x K](bf16)^T
// Wave tile: 16M x 32N (two 16x16 accumulators). Block: 4 waves = 16M x 128N.
// grid.x = (M/16) * (N/128), block = 128 threads.
// ---------------------------------------------------------------------------
__global__ __launch_bounds__(128)
void ptr_gemm_gates(const unsigned short* __restrict__ A,
                    const unsigned short* __restrict__ W,
                    float* __restrict__ out,
                    int M, int N, int K)
{
  const int wave = threadIdx.x >> 5;
  const int lane = threadIdx.x & 31;
  const int ln   = lane & 15;
  const int hi   = lane >> 4;

  const int tilesN = N / 128;
  const int m0 = (blockIdx.x / tilesN) * 16;
  const int n0 = (blockIdx.x % tilesN) * 128 + wave * 32;

  const unsigned short* arow = A + (m0 + ln) * K;
  const unsigned short* w0   = W + (n0 + ln) * K;
  const unsigned short* w1   = W + (n0 + 16 + ln) * K;

  v8f acc0 = {};
  v8f acc1 = {};
  for (int kb = 0; kb < K; kb += 32) {
    FragBF a, b0, b1;
#pragma unroll
    for (int r = 0; r < 8; ++r) {
      // ISA 16-bit A/B layout: VGPR r holds packed K-pair
      // k = (r<4?0:16) + (r%4)*2 + (laneHi? 8:0)
      int k = kb + ((r & 4) << 2) + ((r & 3) << 1) + (hi << 3);
      a.u[r]  = *(const unsigned int*)(arow + k);
      b0.u[r] = *(const unsigned int*)(w0 + k);
      b1.u[r] = *(const unsigned int*)(w1 + k);
    }
    acc0 = __builtin_amdgcn_wmma_f32_16x16x32_bf16(false, a.v, false, b0.v,
                                                   (short)0, acc0, false, false);
    acc1 = __builtin_amdgcn_wmma_f32_16x16x32_bf16(false, a.v, false, b1.v,
                                                   (short)0, acc1, false, false);
  }
#pragma unroll
  for (int r = 0; r < 8; ++r) {
    int mm = m0 + r + hi * 8;      // C/D layout: VGPR r -> row r (+8 for hi half)
    int nn = n0 + ln;
    out[mm * N + nn]      = acc0[r];
    out[mm * N + nn + 16] = acc1[r];
  }
}

// ---------------------------------------------------------------------------
// LSTM pointwise cell. gates[b, 4H] (matmul result); adds x-term + bias,
// applies nonlinearities, updates h/c (fp32), emits bf16 h into the A matrices
// of the next-step GEMMs, optionally records enc_out (bf16) and teacher-forced
// decoder input.
// grid = B blocks, block = H threads.
// ---------------------------------------------------------------------------
__global__ __launch_bounds__(256)
void ptr_lstm_cell(const float* __restrict__ gates,
                   const float* __restrict__ bias,      // [4H]
                   const float* __restrict__ xw,        // x-input weight column
                   int xw_stride,
                   const float* __restrict__ xvals,     // per-batch scalar input
                   int xv_stride,
                   float* __restrict__ h, float* __restrict__ c,
                   unsigned short* __restrict__ A_enc_h,   // [B,256]
                   unsigned short* __restrict__ A_dec_h,   // [B,512] (cols 256..511)
                   unsigned short* __restrict__ enc_out_bf,// [B,S,256] or null
                   int t,
                   const float* __restrict__ x_full,    // [B,S] (F_IN=1)
                   const int* __restrict__ y,           // [B,S]
                   float* __restrict__ dec_next)        // [B] or null
{
  const int b = blockIdx.x;
  const int hid = threadIdx.x;
  const float xb = xvals[b * xv_stride];
  const float* gb = gates + b * NG;

  float gi = gb[hid]            + xb * xw[hid * xw_stride]            + bias[hid];
  float gf = gb[Hsz + hid]      + xb * xw[(Hsz + hid) * xw_stride]    + bias[Hsz + hid];
  float gg = gb[2 * Hsz + hid]  + xb * xw[(2 * Hsz + hid) * xw_stride]+ bias[2 * Hsz + hid];
  float go = gb[3 * Hsz + hid]  + xb * xw[(3 * Hsz + hid) * xw_stride]+ bias[3 * Hsz + hid];

  float i = sigmoidf(gi);
  float f = sigmoidf(gf);
  float g = tanhf(gg);
  float o = sigmoidf(go);
  float cn = f * c[b * Hsz + hid] + i * g;
  float hn = o * tanhf(cn);
  c[b * Hsz + hid] = cn;
  h[b * Hsz + hid] = hn;

  unsigned short hb = f2bf(hn);
  A_enc_h[b * Hsz + hid] = hb;
  A_dec_h[b * 512 + Hsz + hid] = hb;
  if (enc_out_bf) enc_out_bf[(b * Ssz + t) * Hsz + hid] = hb;
  if (dec_next && hid == 0) {
    int yt = y[b * Ssz + t];
    dec_next[b] = x_full[b * Ssz + yt];   // teacher forcing
  }
}

// ---------------------------------------------------------------------------
// enc_proj[b,s,u] = sum_h enc_out[b,s,h] * W1[h,u]   (one-time, small)
// 65536 threads, one per (b,s).
// ---------------------------------------------------------------------------
__global__ __launch_bounds__(256)
void ptr_enc_proj(const unsigned short* __restrict__ enc_out_bf,
                  const float* __restrict__ W1,       // [H,U]
                  float* __restrict__ enc_proj)       // [B,S,U]
{
  int idx = blockIdx.x * 256 + threadIdx.x;           // b*S + s
  const unsigned short* row = enc_out_bf + idx * Hsz;
  float acc[Usz] = {};
  for (int hh = 0; hh < Hsz; ++hh) {
    float v = bf2f(row[hh]);
#pragma unroll
    for (int u = 0; u < Usz; ++u) acc[u] += v * W1[hh * Usz + u];
  }
#pragma unroll
  for (int u = 0; u < Usz; ++u) enc_proj[idx * Usz + u] = acc[u];
}

// ---------------------------------------------------------------------------
// Attention step: scores -> softmax/argmax/CE -> context di (bf16 into A_dec).
// grid = B blocks, 128 threads (s = tid).
// ---------------------------------------------------------------------------
__global__ __launch_bounds__(128)
void ptr_attention(const float* __restrict__ h,          // [B,H]
                   const float* __restrict__ W2,         // [H,U]
                   const float* __restrict__ V,          // [U]
                   const float* __restrict__ enc_proj,   // [B,S,U]
                   const unsigned short* __restrict__ enc_out_bf, // [B,S,H]
                   const int* __restrict__ y,            // [B,S]
                   unsigned short* __restrict__ A_dec,   // [B,512] cols 0..255 = di
                   float* __restrict__ out_preds,        // [S,B] (as float)
                   float* __restrict__ ce_buf,           // [S,B]
                   int t)
{
  const int b = blockIdx.x;
  const int tid = threadIdx.x;                          // 0..127

  __shared__ float hw2[Usz];
  __shared__ float sc[Ssz];
  __shared__ float wts[Ssz];
  __shared__ float redf[Ssz];
  __shared__ int   redi[Ssz];

  if (tid < Usz) {
    float a = 0.f;
    const float* hb = h + b * Hsz;
    for (int k = 0; k < Hsz; ++k) a += hb[k] * W2[k * Usz + tid];
    hw2[tid] = a;
  }
  __syncthreads();

  const float* ep = enc_proj + (b * Ssz + tid) * Usz;
  float s = 0.f;
#pragma unroll
  for (int u = 0; u < Usz; ++u) s += V[u] * tanhf(ep[u] + hw2[u]);
  sc[tid] = s;
  redf[tid] = s;
  redi[tid] = tid;
  __syncthreads();

  // max + first-argmax reduction
  for (int off = 64; off > 0; off >>= 1) {
    if (tid < off && redf[tid + off] > redf[tid]) {
      redf[tid] = redf[tid + off];
      redi[tid] = redi[tid + off];
    }
    __syncthreads();
  }
  const float mx = redf[0];
  const int am = redi[0];
  __syncthreads();

  float e = __expf(s - mx);
  redf[tid] = e;
  __syncthreads();
  for (int off = 64; off > 0; off >>= 1) {
    if (tid < off) redf[tid] += redf[tid + off];
    __syncthreads();
  }
  const float sum = redf[0];
  wts[tid] = e / sum;
  __syncthreads();

  if (tid == 0) {
    out_preds[t * Bsz + b] = (float)am;
    int yt = y[b * Ssz + t];
    float logz = mx + __logf(sum);
    ce_buf[t * Bsz + b] = logz - sc[yt];
  }

  // context di[b,h] = sum_s wts[s] * enc_out[b,s,h]  (coalesced over h)
  const unsigned short* eb = enc_out_bf + b * Ssz * Hsz;
  for (int hh = tid; hh < Hsz; hh += 128) {
    float a = 0.f;
    for (int ss = 0; ss < Ssz; ++ss) a += wts[ss] * bf2f(eb[ss * Hsz + hh]);
    A_dec[b * 512 + hh] = f2bf(a);
  }
}

// Deterministic loss reduction: batch_loss = sum(ce)/B^2
__global__ __launch_bounds__(256)
void ptr_loss_reduce(const float* __restrict__ ce_buf, float* __restrict__ loss)
{
  __shared__ float red[256];
  float a = 0.f;
  for (int i = threadIdx.x; i < Ssz * Bsz; i += 256) a += ce_buf[i];
  red[threadIdx.x] = a;
  __syncthreads();
  for (int off = 128; off > 0; off >>= 1) {
    if (threadIdx.x < off) red[threadIdx.x] += red[threadIdx.x + off];
    __syncthreads();
  }
  if (threadIdx.x == 0) *loss = red[0] * (1.0f / ((float)Bsz * (float)Bsz));
}

// ---------------------------------------------------------------------------
extern "C" void kernel_launch(void* const* d_in, const int* in_sizes, int n_in,
                              void* d_out, int out_size, void* d_ws, size_t ws_size,
                              hipStream_t stream) {
  const float* x      = (const float*)d_in[0];   // [B,S,1]
  const int*   y      = (const int*)  d_in[1];   // [B,S]
  const float* W_ih_e = (const float*)d_in[2];   // [4H,1]
  // const float* W_hh_e used only via convert
  const float* W_hh_e = (const float*)d_in[3];   // [4H,H]
  const float* b_e    = (const float*)d_in[4];   // [4H]
  const float* W_ih_d = (const float*)d_in[5];   // [4H,H+1]
  const float* W_hh_d = (const float*)d_in[6];   // [4H,H]
  const float* b_d    = (const float*)d_in[7];   // [4H]
  const float* W1     = (const float*)d_in[8];   // [H,U]
  const float* W2     = (const float*)d_in[9];   // [H,U]
  const float* V      = (const float*)d_in[10];  // [U]

  // ---- workspace carve (~40 MB total) ----
  char* p = (char*)d_ws;
  float* gates = (float*)p;                 p += (size_t)Bsz * NG * 4;        // 2 MB
  float* h_f   = (float*)p;                 p += (size_t)Bsz * Hsz * 4;       // 512 KB
  float* c_f   = (float*)p;                 p += (size_t)Bsz * Hsz * 4;       // 512 KB
  unsigned short* A_enc = (unsigned short*)p; p += (size_t)Bsz * Hsz * 2;     // 256 KB
  unsigned short* A_dec = (unsigned short*)p; p += (size_t)Bsz * 512 * 2;     // 512 KB
  unsigned short* w_enc = (unsigned short*)p; p += (size_t)NG * Hsz * 2;      // 512 KB
  unsigned short* w_dec = (unsigned short*)p; p += (size_t)NG * 512 * 2;      // 1 MB
  unsigned short* enc_out = (unsigned short*)p; p += (size_t)Bsz * Ssz * Hsz * 2; // 32 MB
  float* enc_proj = (float*)p;              p += (size_t)Bsz * Ssz * Usz * 4; // 2.5 MB
  float* ce_buf   = (float*)p;              p += (size_t)Ssz * Bsz * 4;       // 256 KB
  float* dec_in0  = (float*)p;              p += (size_t)Bsz * 4;
  float* dec_in1  = (float*)p;              p += (size_t)Bsz * 4;

  float* out_preds = (float*)d_out;                 // [S,B]
  float* loss      = (float*)d_out + Ssz * Bsz;     // scalar

  // zero initial state (h0, c0, A_enc<-h0, dec_in0)
  hipMemsetAsync(h_f,   0, (size_t)Bsz * Hsz * 4, stream);
  hipMemsetAsync(c_f,   0, (size_t)Bsz * Hsz * 4, stream);
  hipMemsetAsync(A_enc, 0, (size_t)Bsz * Hsz * 2, stream);
  hipMemsetAsync(dec_in0, 0, (size_t)Bsz * 4, stream);
  hipMemsetAsync(dec_in1, 0, (size_t)Bsz * 4, stream);

  ptr_convert_weights<<<(NG * 512 + 255) / 256, 256, 0, stream>>>(
      W_hh_e, W_ih_d, W_hh_d, w_enc, w_dec);

  const int gemm_grid = (Bsz / 16) * (NG / 128);    // 256 blocks

  // ---- encoder ----
  for (int t = 0; t < Ssz; ++t) {
    ptr_gemm_gates<<<gemm_grid, 128, 0, stream>>>(A_enc, w_enc, gates,
                                                  Bsz, NG, Hsz);
    ptr_lstm_cell<<<Bsz, Hsz, 0, stream>>>(
        gates, b_e,
        /*xw=*/W_ih_e, /*xw_stride=*/1,
        /*xvals=*/x + t, /*xv_stride=*/Ssz,
        h_f, c_f, A_enc, A_dec, enc_out, t,
        /*x_full=*/nullptr, /*y=*/nullptr, /*dec_next=*/nullptr);
  }

  ptr_enc_proj<<<(Bsz * Ssz) / 256, 256, 0, stream>>>(enc_out, W1, enc_proj);

  // ---- decoder ----
  for (int t = 0; t < Ssz; ++t) {
    float* din_cur = (t & 1) ? dec_in1 : dec_in0;
    float* din_nxt = (t & 1) ? dec_in0 : dec_in1;
    ptr_attention<<<Bsz, 128, 0, stream>>>(
        h_f, W2, V, enc_proj, enc_out, y, A_dec, out_preds, ce_buf, t);
    ptr_gemm_gates<<<gemm_grid, 128, 0, stream>>>(A_dec, w_dec, gates,
                                                  Bsz, NG, 512);
    ptr_lstm_cell<<<Bsz, Hsz, 0, stream>>>(
        gates, b_d,
        /*xw=*/W_ih_d + Hsz, /*xw_stride=*/Hsz + 1,
        /*xvals=*/din_cur, /*xv_stride=*/1,
        h_f, c_f, A_enc, A_dec, /*enc_out=*/nullptr, t,
        /*x_full=*/x, /*y=*/y, /*dec_next=*/din_nxt);
  }

  ptr_loss_reduce<<<1, 256, 0, stream>>>(ce_buf, loss);
}
